// SAGPool_11218454577330
// MI455X (gfx1250) — matrix operand order repaired and checked by reading, not compiled, not run on hardware
//
#include <hip/hip_runtime.h>
#include <hip/hip_bf16.h>
#include <stdint.h>

// ---------------------------------------------------------------------------
// Problem constants (from reference)
// ---------------------------------------------------------------------------
static constexpr int GG   = 100;       // graphs
static constexpr int NPER = 1000;      // nodes per graph (layer 0)
static constexpr int NN0  = GG * NPER; // 100000 nodes
static constexpr int EE   = GG * 16000;// 1.6M edges
static constexpr int EDIM = 16;
#define GEN_EPS 1e-7f

typedef __attribute__((ext_vector_type(16))) _Float16 v16h;
typedef __attribute__((ext_vector_type(8)))  _Float16 v8h;
typedef __attribute__((ext_vector_type(8)))  float    v8f;

// ---------------------------------------------------------------------------
// f32 -> f16 conversion (activations), and weight transpose+convert:
//   Wt[n*K + k] = (f16) W[k*N + n]
// ---------------------------------------------------------------------------
__global__ void cvt_f16_kernel(const float* __restrict__ in, _Float16* __restrict__ out,
                               long long n)
{
    long long i = (long long)blockIdx.x * blockDim.x + threadIdx.x;
    if (i < n) out[i] = (_Float16)in[i];
}

__global__ void transpose_w_kernel(const float* __restrict__ W, _Float16* __restrict__ Wt,
                                   int K, int N)
{
    int t = blockIdx.x * blockDim.x + threadIdx.x;
    if (t >= K * N) return;
    int n = t % N, k = t / N;
    Wt[(size_t)n * K + k] = (_Float16)W[(size_t)k * N + n];
}

// ---------------------------------------------------------------------------
// WMMA GEMM: C[M,N] = Ah[M,K](f16) @ Wt[N,K](f16, pre-transposed) + bias.
// One wave32 per 16x16 output tile. K must be a multiple of 32 (true for all
// call sites: 64/128/256). Inner loop: 4x b128 loads + 1x v_wmma.
// Fragment layouts per CDNA5 ISA 7.12.2:
//   A 16x32 f16 : lane&15 = row M, K = e + (e>=8?8:0) + (lane>=16?8:0)
//                 -> per lane two contiguous 8-half chunks at +0 and +16
//   B 32x16 f16 : lane&15 = col N, K = e + (e>=8?8:0) + (lane>=16?8:0)
//                 -> per lane two contiguous 8-half chunks at +0 and +16
//   C/D 16x16 f32: elem r -> row = r + (lane>=16?8:0), col = lane&15
// Tile-edge rows/cols: loads are clamped (branch-free), stores are guarded.
// ---------------------------------------------------------------------------
__global__ void gemm_wmma_kernel(const _Float16* __restrict__ Ah,
                                 const _Float16* __restrict__ Wt,
                                 const float* __restrict__ bias,
                                 float* __restrict__ C,
                                 int M, int K, int N, int fuse_relu)
{
    const int lane = threadIdx.x & 31;
    const int wave = blockIdx.x * (blockDim.x >> 5) + (threadIdx.x >> 5);
    const int tilesN = (N + 15) >> 4;
    const int tilesM = (M + 15) >> 4;
    if (wave >= tilesM * tilesN) return;
    const int tm = wave / tilesN, tn = wave % tilesN;
    const int row0 = tm << 4, col0 = tn << 4;

    int rA = row0 + (lane & 15); rA = (rA < M) ? rA : (M - 1);   // clamp loads
    int cB = col0 + (lane & 15); cB = (cB < N) ? cB : (N - 1);
    const _Float16* __restrict__ arow = Ah + (size_t)rA * K + ((lane >> 4) << 3);
    const _Float16* __restrict__ brow = Wt + (size_t)cB * K + ((lane >> 4) << 3);

    v8f acc = {};
#pragma unroll 2
    for (int k0 = 0; k0 < K; k0 += 32) {
        v8h a0 = *(const v8h*)(arow + k0);
        v8h a1 = *(const v8h*)(arow + k0 + 16);
        v8h b0 = *(const v8h*)(brow + k0);
        v8h b1 = *(const v8h*)(brow + k0 + 16);
        v16h a = __builtin_shufflevector(a0, a1, 0,1,2,3,4,5,6,7,8,9,10,11,12,13,14,15);
        v16h b = __builtin_shufflevector(b0, b1, 0,1,2,3,4,5,6,7,8,9,10,11,12,13,14,15);
        acc = __builtin_amdgcn_wmma_f32_16x16x32_f16(false, a, false, b,
                                                     (short)0, acc, false, false);
    }

    const int rofs = (lane >> 4) << 3;
    const int col  = col0 + (lane & 15);
#pragma unroll
    for (int r = 0; r < 8; ++r) {
        int row = row0 + r + rofs;
        if (row < M && col < N) {
            float v = acc[r] + (bias ? bias[col] : 0.0f);
            if (fuse_relu) v = fmaxf(v, 0.0f);
            C[(size_t)row * N + col] = v;
        }
    }
}

// ---------------------------------------------------------------------------
// Edge message helpers: e_c = be[c] + sum_k edge_attr[eid,k]*We[k,c] (K=16)
// msg = relu(h_src[src,c] + e_c) + eps  (always > 0)
// ---------------------------------------------------------------------------
__device__ __forceinline__ float edge_proj(const float* __restrict__ ea,
                                           const float* __restrict__ We,
                                           const float* __restrict__ be,
                                           int eid, int c, int C)
{
    const float* row = ea + (size_t)eid * EDIM;
    float acc = be[c];
#pragma unroll
    for (int k = 0; k < EDIM; ++k) acc += row[k] * We[k * C + c];
    return acc;
}

__global__ void edge_max_kernel(const float* __restrict__ hsrc,
                                const float* __restrict__ ea,
                                const float* __restrict__ We,
                                const float* __restrict__ be,
                                const int* __restrict__ src,
                                const int* __restrict__ dst,
                                const unsigned char* __restrict__ mask,
                                float* __restrict__ m, int C)
{
    long long t = (long long)blockIdx.x * blockDim.x + threadIdx.x;
    if (t >= (long long)EE * C) return;
    int eid = (int)(t / C), c = (int)(t % C);
    if (!mask[eid]) return;
    float msg = fmaxf(hsrc[(size_t)src[eid] * C + c] + edge_proj(ea, We, be, eid, c, C), 0.0f) + GEN_EPS;
    atomicMax((int*)&m[(size_t)dst[eid] * C + c], __float_as_int(msg)); // msg>0: int order == float order
}

__global__ void edge_sum_kernel(const float* __restrict__ hsrc,
                                const float* __restrict__ ea,
                                const float* __restrict__ We,
                                const float* __restrict__ be,
                                const int* __restrict__ src,
                                const int* __restrict__ dst,
                                const unsigned char* __restrict__ mask,
                                const float* __restrict__ m,
                                float* __restrict__ s, int C)
{
    long long t = (long long)blockIdx.x * blockDim.x + threadIdx.x;
    if (t >= (long long)EE * C) return;
    int eid = (int)(t / C), c = (int)(t % C);
    if (!mask[eid]) return;
    size_t di = (size_t)dst[eid] * C + c;
    float msg = fmaxf(hsrc[(size_t)src[eid] * C + c] + edge_proj(ea, We, be, eid, c, C), 0.0f) + GEN_EPS;
    atomicAdd(&s[di], __expf(msg - m[di]));
}

__global__ void edge_aggr_kernel(const float* __restrict__ hsrc,
                                 const float* __restrict__ ea,
                                 const float* __restrict__ We,
                                 const float* __restrict__ be,
                                 const int* __restrict__ src,
                                 const int* __restrict__ dst,
                                 const unsigned char* __restrict__ mask,
                                 const float* __restrict__ m,
                                 const float* __restrict__ s,
                                 float* __restrict__ aggr, int C)
{
    long long t = (long long)blockIdx.x * blockDim.x + threadIdx.x;
    if (t >= (long long)EE * C) return;
    int eid = (int)(t / C), c = (int)(t % C);
    if (!mask[eid]) return;
    size_t di = (size_t)dst[eid] * C + c;
    float msg = fmaxf(hsrc[(size_t)src[eid] * C + c] + edge_proj(ea, We, be, eid, c, C), 0.0f) + GEN_EPS;
    float w = __expf(msg - m[di]);
    atomicAdd(&aggr[di], (w / fmaxf(s[di], 1e-16f)) * msg);
}

// ---------------------------------------------------------------------------
// Elementwise / reduction kernels
// ---------------------------------------------------------------------------
__global__ void add_kernel(float* a, const float* b, long long n)
{
    long long i = (long long)blockIdx.x * blockDim.x + threadIdx.x;
    if (i < n) a[i] += b[i];
}

__global__ void relu_kernel(float* a, long long n)
{
    long long i = (long long)blockIdx.x * blockDim.x + threadIdx.x;
    if (i < n) a[i] = fmaxf(a[i], 0.0f);
}

__global__ void bn_stats_kernel(const float* __restrict__ h, int n, int cols,
                                float* __restrict__ mu, float* __restrict__ var)
{
    int c = blockIdx.x;
    __shared__ float ssum[256], ssq[256];
    float s = 0.0f, q = 0.0f;
    for (int i = threadIdx.x; i < n; i += blockDim.x) {
        float v = h[(size_t)i * cols + c];
        s += v; q += v * v;
    }
    ssum[threadIdx.x] = s; ssq[threadIdx.x] = q;
    __syncthreads();
    for (int o = 128; o > 0; o >>= 1) {
        if (threadIdx.x < o) { ssum[threadIdx.x] += ssum[threadIdx.x + o]; ssq[threadIdx.x] += ssq[threadIdx.x + o]; }
        __syncthreads();
    }
    if (threadIdx.x == 0) {
        float mean = ssum[0] / n;
        mu[c] = mean;
        var[c] = ssq[0] / n - mean * mean;
    }
}

__global__ void bn_apply_relu_kernel(float* h, const float* mu, const float* var,
                                     const float* gamma, const float* beta,
                                     long long total, int cols)
{
    long long i = (long long)blockIdx.x * blockDim.x + threadIdx.x;
    if (i >= total) return;
    int c = (int)(i % cols);
    float v = gamma[c] * (h[i] - mu[c]) * rsqrtf(var[c] + 1e-5f) + beta[c];
    h[i] = fmaxf(v, 0.0f);
}

// ---------------------------------------------------------------------------
// SAGPool kernels
// ---------------------------------------------------------------------------
__global__ void node_dot_kernel(const float* __restrict__ x, const float* __restrict__ w,
                                float* __restrict__ out, int n, int C)
{
    int i = blockIdx.x * blockDim.x + threadIdx.x;
    if (i >= n) return;
    float acc = 0.0f;
    for (int c = 0; c < C; ++c) acc += x[(size_t)i * C + c] * w[c];
    out[i] = acc;
}

__global__ void edge_rel_agg_kernel(const float* __restrict__ rel,
                                    const int* __restrict__ src,
                                    const int* __restrict__ dst,
                                    const unsigned char* __restrict__ mask,
                                    float* __restrict__ agg)
{
    int i = blockIdx.x * blockDim.x + threadIdx.x;
    if (i >= EE || !mask[i]) return;
    atomicAdd(&agg[dst[i]], rel[src[i]]);
}

__global__ void score_kernel(const float* agg, const float* rootd,
                             const float* brel, float* score, int n)
{
    int i = blockIdx.x * blockDim.x + threadIdx.x;
    if (i >= n) return;
    score[i] = tanhf(agg[i] + brel[0] + rootd[i]);
}

// one block per graph; in-LDS bitonic sort of (score, ~idx) keys, descending.
__global__ void topk_kernel(const float* __restrict__ score, int n_per, int npow2,
                            int k, int* __restrict__ perm)
{
    extern __shared__ unsigned long long keys[];
    int g = blockIdx.x;
    const float* sg = score + (size_t)g * n_per;
    for (int i = threadIdx.x; i < npow2; i += blockDim.x) {
        unsigned long long key = 0ull; // pad = smallest
        if (i < n_per) {
            unsigned u = __float_as_uint(sg[i]);
            u = (u & 0x80000000u) ? ~u : (u | 0x80000000u); // order-preserving map
            key = ((unsigned long long)u << 32) | (unsigned)(0xFFFFFFFFu - (unsigned)i);
        }
        keys[i] = key;
    }
    __syncthreads();
    for (int ks = 2; ks <= npow2; ks <<= 1) {
        for (int j = ks >> 1; j > 0; j >>= 1) {
            for (int i = threadIdx.x; i < npow2; i += blockDim.x) {
                int l = i ^ j;
                if (l > i) {
                    unsigned long long a = keys[i], b = keys[l];
                    bool dirDesc = ((i & ks) == 0);
                    if (dirDesc ? (a < b) : (a > b)) { keys[i] = b; keys[l] = a; }
                }
            }
            __syncthreads();
        }
    }
    for (int i = threadIdx.x; i < k; i += blockDim.x) {
        unsigned idx_local = 0xFFFFFFFFu - (unsigned)(keys[i] & 0xFFFFFFFFull);
        perm[g * k + i] = g * n_per + (int)idx_local;
    }
}

__global__ void set_newid_kernel(const int* perm, int* new_id, int nk)
{
    int i = blockIdx.x * blockDim.x + threadIdx.x;
    if (i < nk) new_id[perm[i]] = i;
}

__global__ void gather_scale_kernel(const float* __restrict__ x, const float* __restrict__ score,
                                    const int* __restrict__ perm, float* __restrict__ xn,
                                    int nk, int C)
{
    long long t = (long long)blockIdx.x * blockDim.x + threadIdx.x;
    if (t >= (long long)nk * C) return;
    int j = (int)(t / C), c = (int)(t % C);
    int p = perm[j];
    xn[t] = x[(size_t)p * C + c] * score[p];
}

__global__ void remap_edges_kernel(int* src, int* dst, unsigned char* mask,
                                   const int* __restrict__ new_id)
{
    int i = blockIdx.x * blockDim.x + threadIdx.x;
    if (i >= EE) return;
    int ns = new_id[src[i]], nd = new_id[dst[i]];
    bool valid = mask[i] && ns >= 0 && nd >= 0;
    src[i] = valid ? ns : 0;
    dst[i] = valid ? nd : 0;
    mask[i] = valid ? 1 : 0;
}

// ---------------------------------------------------------------------------
// Head kernels
// ---------------------------------------------------------------------------
__global__ void meanpool_kernel(const float* __restrict__ h, float* __restrict__ pooled, int C)
{
    int t = blockIdx.x * blockDim.x + threadIdx.x;
    if (t >= GG * C) return;
    int g = t / C, c = t % C;
    float s = 0.0f;
#pragma unroll
    for (int j = 0; j < 8; ++j) s += h[(size_t)(g * 8 + j) * C + c];
    pooled[t] = s * (1.0f / 8.0f);
}

__global__ void log_softmax_kernel(const float* __restrict__ z, float* __restrict__ out, int D)
{
    int g = blockIdx.x * blockDim.x + threadIdx.x;
    if (g >= GG) return;
    float mx = -3.0e38f;
    for (int d = 0; d < D; ++d) mx = fmaxf(mx, z[g * D + d]);
    float s = 0.0f;
    for (int d = 0; d < D; ++d) s += __expf(z[g * D + d] - mx);
    float l = __logf(s);
    for (int d = 0; d < D; ++d) out[g * D + d] = z[g * D + d] - mx - l;
}

// ---------------------------------------------------------------------------
// Host orchestration
// ---------------------------------------------------------------------------
static inline unsigned div_up_ll(long long n, int b) { return (unsigned)((n + b - 1) / b); }

// scratch for f16 operands (set once per kernel_launch call from d_ws)
static _Float16* g_ah = nullptr;   // converted activations [M,K] f16
static _Float16* g_wt = nullptr;   // transposed+converted weights [N,K] f16

// C = A @ W + bias. cvtA=false reuses g_ah from the previous call (same A,K).
static void gemm(hipStream_t st, const float* A, const float* W, const float* bias,
                 float* C, int M, int K, int N, int relu, bool cvtA)
{
    if (cvtA)
        cvt_f16_kernel<<<div_up_ll((long long)M * K, 256), 256, 0, st>>>(A, g_ah, (long long)M * K);
    transpose_w_kernel<<<div_up_ll((long long)K * N, 256), 256, 0, st>>>(W, g_wt, K, N);
    int tiles = ((M + 15) / 16) * ((N + 15) / 16);
    int blocks = (tiles + 7) / 8;                 // 8 waves (256 threads) per block
    gemm_wmma_kernel<<<blocks, 256, 0, st>>>(g_ah, g_wt, bias, C, M, K, N, relu);
}

static void run_genconv(hipStream_t st, const float* x, int n, int Cin, int C,
                        const float* Wsrc, const float* bsrc, const float* Wdst, const float* bdst,
                        const float* We, const float* be, const float* W1, const float* b1,
                        const float* gamma, const float* beta, const float* W2, const float* b2,
                        const float* edge_attr, const int* src, const int* dst,
                        const unsigned char* mask,
                        float* hsrcB, float* hdstB, float* mB, float* sB, float* agB,
                        float* h1B, float* muB, float* varB, float* outB)
{
    const float* hsrc = x;
    const float* hdst = x;
    if (Wsrc) {
        gemm(st, x, Wsrc, bsrc, hsrcB, n, Cin, C, /*relu=*/0, /*cvtA=*/true);
        gemm(st, x, Wdst, bdst, hdstB, n, Cin, C, /*relu=*/0, /*cvtA=*/false); // reuse converted x
        hsrc = hsrcB; hdst = hdstB;
    }
    size_t nC = (size_t)n * C;
    long long totE = (long long)EE * C;
    (void)hipMemsetAsync(mB, 0, nC * 4, st);
    edge_max_kernel<<<div_up_ll(totE, 256), 256, 0, st>>>(hsrc, edge_attr, We, be, src, dst, mask, mB, C);
    (void)hipMemsetAsync(sB, 0, nC * 4, st);
    edge_sum_kernel<<<div_up_ll(totE, 256), 256, 0, st>>>(hsrc, edge_attr, We, be, src, dst, mask, mB, sB, C);
    (void)hipMemsetAsync(agB, 0, nC * 4, st);
    edge_aggr_kernel<<<div_up_ll(totE, 256), 256, 0, st>>>(hsrc, edge_attr, We, be, src, dst, mask, mB, sB, agB, C);
    add_kernel<<<div_up_ll((long long)nC, 256), 256, 0, st>>>(agB, hdst, (long long)nC);
    gemm(st, agB, W1, b1, h1B, n, C, 2 * C, 0, true);
    bn_stats_kernel<<<2 * C, 256, 0, st>>>(h1B, n, 2 * C, muB, varB);
    bn_apply_relu_kernel<<<div_up_ll((long long)n * 2 * C, 256), 256, 0, st>>>(h1B, muB, varB, gamma, beta, (long long)n * 2 * C, 2 * C);
    gemm(st, h1B, W2, b2, outB, n, 2 * C, C, 0, true);
}

static void run_sagpool(hipStream_t st, const float* x, int n, int C,
                        const float* Wrel, const float* brel, const float* Wroot,
                        int n_per, int k,
                        int* src, int* dst, unsigned char* mask,
                        float* relB, float* aggB, float* rootB, float* scoreB,
                        int* perm, int* new_id, float* xnew)
{
    node_dot_kernel<<<(n + 255) / 256, 256, 0, st>>>(x, Wrel, relB, n, C);
    (void)hipMemsetAsync(aggB, 0, (size_t)n * 4, st);
    edge_rel_agg_kernel<<<(EE + 255) / 256, 256, 0, st>>>(relB, src, dst, mask, aggB);
    node_dot_kernel<<<(n + 255) / 256, 256, 0, st>>>(x, Wroot, rootB, n, C);
    score_kernel<<<(n + 255) / 256, 256, 0, st>>>(aggB, rootB, brel, scoreB, n);
    int npow2 = 1; while (npow2 < n_per) npow2 <<= 1;
    topk_kernel<<<GG, 256, (size_t)npow2 * sizeof(unsigned long long), st>>>(scoreB, n_per, npow2, k, perm);
    (void)hipMemsetAsync(new_id, 0xFF, (size_t)n * 4, st);      // -1
    int nk = GG * k;
    set_newid_kernel<<<(nk + 255) / 256, 256, 0, st>>>(perm, new_id, nk);
    gather_scale_kernel<<<div_up_ll((long long)nk * C, 256), 256, 0, st>>>(x, scoreB, perm, xnew, nk, C);
    remap_edges_kernel<<<(EE + 255) / 256, 256, 0, st>>>(src, dst, mask, new_id);
}

extern "C" void kernel_launch(void* const* d_in, const int* in_sizes, int n_in,
                              void* d_out, int out_size, void* d_ws, size_t ws_size,
                              hipStream_t stream)
{
    (void)in_sizes; (void)n_in; (void)out_size; (void)ws_size;

    // ---- input unpacking (dict order of setup_inputs) ----
    int ii = 0;
    auto F = [&](void) { return (const float*)d_in[ii++]; };
    const float* x         = F();
    const float* edge_attr = F();
    // pc1 (in=128, out=64, has_lin)
    const float *Wsrc1=F(), *bsrc1=F(), *Wdst1=F(), *bdst1=F(), *We1=F(), *be1=F(),
                *W11=F(), *b11=F(), *g1=F(), *bt1=F(), *W21=F(), *b21=F();
    // pp1
    const float *Wrel1=F(), *brel1=F(), *Wroot1=F();
    // pc2 (64->64, no lin)
    const float *We2=F(), *be2=F(), *W12=F(), *b12=F(), *g2=F(), *bt2=F(), *W22=F(), *b22=F();
    // pp2
    const float *Wrel2=F(), *brel2=F(), *Wroot2=F();
    // pc3 (64->128, has_lin)
    const float *Wsrc3=F(), *bsrc3=F(), *Wdst3=F(), *bdst3=F(), *We3=F(), *be3=F(),
                *W13=F(), *b13=F(), *g3=F(), *bt3=F(), *W23=F(), *b23=F();
    // pp3
    const float *Wrel3=F(), *brel3=F(), *Wroot3=F();
    // head
    const float *hW1=F(), *hb1=F(), *hW2=F(), *hb2=F();
    const int* edge_index = (const int*)d_in[ii++];
    // (batch input unused: node ordering is graph-major by construction)

    // ---- workspace bump allocator ----
    char* p = (char*)d_ws;
    auto take = [&](size_t bytes) -> void* {
        void* r = (void*)p;
        p += (bytes + 255) & ~(size_t)255;
        return r;
    };
    int*           srcW  = (int*)take((size_t)EE * 4);
    int*           dstW  = (int*)take((size_t)EE * 4);
    unsigned char* maskW = (unsigned char*)take((size_t)EE);
    float* hsrcB = (float*)take((size_t)NN0 * 64 * 4);
    float* hdstB = (float*)take((size_t)NN0 * 64 * 4);
    float* mB    = (float*)take((size_t)NN0 * 64 * 4);
    float* sB    = (float*)take((size_t)NN0 * 64 * 4);
    float* agB   = (float*)take((size_t)NN0 * 64 * 4);
    float* h1B   = (float*)take((size_t)NN0 * 128 * 4);
    float* houtB = (float*)take((size_t)NN0 * 64 * 4);
    float* xpA   = (float*)take((size_t)20000 * 128 * 4);
    float* xpB   = (float*)take((size_t)4000 * 128 * 4);
    float* relB   = (float*)take((size_t)NN0 * 4);
    float* aggsB  = (float*)take((size_t)NN0 * 4);
    float* rootB  = (float*)take((size_t)NN0 * 4);
    float* scoreB = (float*)take((size_t)NN0 * 4);
    int*   permB  = (int*)take((size_t)20000 * 4);
    int*   newidB = (int*)take((size_t)NN0 * 4);
    float* muB  = (float*)take(256 * 4);
    float* varB = (float*)take(256 * 4);
    float* pooledB = (float*)take((size_t)GG * 128 * 4);
    float* z1B     = (float*)take((size_t)GG * 64 * 4);
    float* zB      = (float*)take((size_t)GG * 10 * 4);
    g_ah = (_Float16*)take((size_t)NN0 * 128 * 2);   // f16 activations (max M*K)
    g_wt = (_Float16*)take((size_t)256 * 256 * 2);   // f16 transposed weights (max K*N)

    // ---- init working edge lists (fresh every call: deterministic) ----
    (void)hipMemcpyAsync(srcW, edge_index,      (size_t)EE * 4, hipMemcpyDeviceToDevice, stream);
    (void)hipMemcpyAsync(dstW, edge_index + EE, (size_t)EE * 4, hipMemcpyDeviceToDevice, stream);
    (void)hipMemsetAsync(maskW, 1, (size_t)EE, stream);

    // ---- layer 1: GENConv(128->64) + SAGPool(1000->200) ----
    run_genconv(stream, x, NN0, 128, 64,
                Wsrc1, bsrc1, Wdst1, bdst1, We1, be1, W11, b11, g1, bt1, W21, b21,
                edge_attr, srcW, dstW, maskW,
                hsrcB, hdstB, mB, sB, agB, h1B, muB, varB, houtB);
    run_sagpool(stream, houtB, NN0, 64, Wrel1, brel1, Wroot1, NPER, 200,
                srcW, dstW, maskW, relB, aggsB, rootB, scoreB, permB, newidB, xpA);
    relu_kernel<<<div_up_ll((long long)20000 * 64, 256), 256, 0, stream>>>(xpA, (long long)20000 * 64);

    // ---- layer 2: GENConv(64->64, no lin) + SAGPool(200->40) ----
    run_genconv(stream, xpA, 20000, 64, 64,
                nullptr, nullptr, nullptr, nullptr, We2, be2, W12, b12, g2, bt2, W22, b22,
                edge_attr, srcW, dstW, maskW,
                hsrcB, hdstB, mB, sB, agB, h1B, muB, varB, houtB);
    run_sagpool(stream, houtB, 20000, 64, Wrel2, brel2, Wroot2, 200, 40,
                srcW, dstW, maskW, relB, aggsB, rootB, scoreB, permB, newidB, xpB);
    relu_kernel<<<div_up_ll((long long)4000 * 64, 256), 256, 0, stream>>>(xpB, (long long)4000 * 64);

    // ---- layer 3: GENConv(64->128) + SAGPool(40->8) ----
    run_genconv(stream, xpB, 4000, 64, 128,
                Wsrc3, bsrc3, Wdst3, bdst3, We3, be3, W13, b13, g3, bt3, W23, b23,
                edge_attr, srcW, dstW, maskW,
                hsrcB, hdstB, mB, sB, agB, h1B, muB, varB, houtB);
    run_sagpool(stream, houtB, 4000, 128, Wrel3, brel3, Wroot3, 40, 8,
                srcW, dstW, maskW, relB, aggsB, rootB, scoreB, permB, newidB, xpA);
    relu_kernel<<<div_up_ll((long long)800 * 128, 256), 256, 0, stream>>>(xpA, (long long)800 * 128);

    // ---- head: global mean pool + 2-layer MLP + log_softmax ----
    meanpool_kernel<<<(GG * 128 + 255) / 256, 256, 0, stream>>>(xpA, pooledB, 128);
    gemm(stream, pooledB, hW1, hb1, z1B, GG, 128, 64, 0, true);
    gemm(stream, z1B, hW2, hb2, zB, GG, 64, 10, 0, true);
    log_softmax_kernel<<<1, 128, 0, stream>>>(zB, (float*)d_out, 10);
}
